// TransformerEncoder_17712445129131
// MI455X (gfx1250) — matrix operand order repaired and checked by reading, not compiled, not run on hardware
//
#include <hip/hip_runtime.h>

// ---------------------------------------------------------------- types
using bf16 = __bf16;
typedef __attribute__((ext_vector_type(16))) __bf16 v16bf;
typedef __attribute__((ext_vector_type(8)))  __bf16 v8bf;
typedef __attribute__((ext_vector_type(4)))  __bf16 v4bf;
typedef __attribute__((ext_vector_type(8)))  float  v8f;
typedef __attribute__((ext_vector_type(4)))  int    v4i;

__device__ __forceinline__ v8f wmma_bf16(v16bf a, v16bf b, v8f c) {
  return __builtin_amdgcn_wmma_f32_16x16x32_bf16(
      /*neg_a=*/false, a, /*neg_b=*/false, b,
      /*c_mod=*/(short)0, c, /*reuse_a=*/false, /*reuse_b=*/false);
}

__device__ __forceinline__ v16bf loadAB(const bf16* plo, const bf16* phi) {
  v8bf lo = *(const v8bf*)plo;
  v8bf hi = *(const v8bf*)phi;
  return __builtin_shufflevector(lo, hi, 0,1,2,3,4,5,6,7,8,9,10,11,12,13,14,15);
}

// optional CDNA5 async global->LDS path (ASYNCcnt)
#if defined(__has_builtin)
#  if __has_builtin(__builtin_amdgcn_global_load_async_to_lds_b128) && \
      __has_builtin(__builtin_amdgcn_s_wait_asynccnt)
#    define HAVE_ASYNC_LDS 1
#  endif
#endif
#ifndef HAVE_ASYNC_LDS
#  define HAVE_ASYNC_LDS 0
#endif

#if HAVE_ASYNC_LDS
typedef __attribute__((address_space(1))) v4i* gv4i_p;   // global int4*
typedef __attribute__((address_space(3))) v4i* lv4i_p;   // LDS int4*
__device__ __forceinline__ void async_cp16(bf16* lds, const bf16* g) {
  bf16* gg = const_cast<bf16*>(g);
  __builtin_amdgcn_global_load_async_to_lds_b128(
      (gv4i_p)gg, (lv4i_p)lds, /*offset=*/0, /*cpol=*/0);
}
#endif

// ---------------------------------------------------------------- sizes
constexpr int BB = 32, NN = 512, NDIM = 8, LL = 4, HH = 8, DD = 512, DK = 64, FF = 2048;
constexpr size_t HB = (size_t)BB * NN;   // 16384 tokens

// workspace layout (bytes)
constexpr size_t OFF_H     = 0;
constexpr size_t OFF_HRES  = OFF_H     + HB * DD * 4;
constexpr size_t OFF_HBF   = OFF_HRES  + HB * DD * 4;
constexpr size_t OFF_QBF   = OFF_HBF   + HB * DD * 2;
constexpr size_t OFF_KBF   = OFF_QBF   + HB * DD * 2;
constexpr size_t OFF_VBF   = OFF_KBF   + HB * DD * 2;
constexpr size_t OFF_HEADS = OFF_VBF   + HB * DD * 2;
constexpr size_t OFF_FFBF  = OFF_HEADS + HB * DD * 2;
constexpr size_t OFF_WQT   = OFF_FFBF  + HB * FF * 2;
constexpr size_t OFF_WKT   = OFF_WQT   + (size_t)DD * DD * 2;
constexpr size_t OFF_WVT   = OFF_WKT   + (size_t)DD * DD * 2;
constexpr size_t OFF_WOT   = OFF_WVT   + (size_t)DD * DD * 2;
constexpr size_t OFF_W1T   = OFF_WOT   + (size_t)DD * DD * 2;
constexpr size_t OFF_W2T   = OFF_W1T   + (size_t)DD * FF * 2;
constexpr size_t OFF_MU    = OFF_W2T   + (size_t)DD * FF * 2;
constexpr size_t OFF_RINV  = OFF_MU    + DD * 4;
constexpr size_t OFF_MB    = OFF_RINV  + DD * 4;                 // u64 bitmask [B,N,8]

// ---------------------------------------------------------------- weight repack (f32 -> bf16, B^T layout [N][K])
__global__ void pack_qkvT(const float* __restrict__ W, bf16* __restrict__ out) {
  int i = blockIdx.x * 256 + threadIdx.x;          // H*D*DK = 262144
  int c = i >> 9, d = i & 511;
  int h = c >> 6, k = c & 63;
  out[i] = (bf16)W[((size_t)h * DD + d) * DK + k];
}
__global__ void pack_woT(const float* __restrict__ W, bf16* __restrict__ out) {
  int i = blockIdx.x * 256 + threadIdx.x;
  int d = i >> 9, c = i & 511;
  int h = c >> 6, k = c & 63;
  out[i] = (bf16)W[((size_t)h * DK + k) * DD + d];
}
__global__ void pack_w1T(const float* __restrict__ W, bf16* __restrict__ out) {
  int i = blockIdx.x * 256 + threadIdx.x;          // D*F = 1048576
  int f = i >> 9, d = i & 511;
  out[i] = (bf16)W[(size_t)d * FF + f];
}
__global__ void pack_w2T(const float* __restrict__ W, bf16* __restrict__ out) {
  int i = blockIdx.x * 256 + threadIdx.x;
  int d = i >> 11, f = i & 2047;
  out[i] = (bf16)W[(size_t)f * DD + d];
}
// mask [B,N,N] int -> 512-bit rows (8 x u64 per row)
__global__ void pack_mask(const int* __restrict__ mask, unsigned long long* __restrict__ mb) {
  int i = blockIdx.x * 256 + threadIdx.x;          // B*N*8 = 131072 words
  const int* src = mask + (size_t)i * 64;
  unsigned long long w = 0ull;
#pragma unroll 8
  for (int j = 0; j < 64; ++j) w |= (unsigned long long)(src[j] != 0) << j;
  mb[i] = w;
}

// ---------------------------------------------------------------- init embed: h = x @ We + be (K=8)
__global__ void init_embed(const float* __restrict__ x, const float* __restrict__ We,
                           const float* __restrict__ be, float* __restrict__ H,
                           bf16* __restrict__ Hb) {
  size_t i4 = ((size_t)blockIdx.x * 256 + threadIdx.x) * 4;   // HB*D elems
  int d = (int)(i4 & 511);
  size_t bn = i4 >> 9;
  float4 s = *(const float4*)&be[d];
#pragma unroll
  for (int j = 0; j < NDIM; ++j) {
    float xv = x[bn * NDIM + j];
    const float4 w = *(const float4*)&We[j * DD + d];
    s.x += xv * w.x; s.y += xv * w.y; s.z += xv * w.z; s.w += xv * w.w;
  }
  *(float4*)&H[i4] = s;
  v4bf o = {(bf16)s.x, (bf16)s.y, (bf16)s.z, (bf16)s.w};
  *(v4bf*)&Hb[i4] = o;
}

// ---------------------------------------------------------------- WMMA GEMM: C[M,N] = A[M,K](bf16) * Bt[N,K]^T(bf16)
// block tile 128x256, wave tile 64x64 (4x4 frags), K-step 32, double-buffered LDS
// MODE 0: ->bf16   1: +resid ->f32   2: +bias,relu ->bf16   3: +bias +resid ->f32
template <int MODE>
__global__ __launch_bounds__(256) void gemm_bf16(
    const bf16* __restrict__ A, const bf16* __restrict__ Bt,
    const float* __restrict__ bias, const float* __restrict__ resid,
    float* __restrict__ Cf, bf16* __restrict__ Cb, int M, int N, int K) {
  __shared__ bf16 sA[2][128][40];   // 80B rows, 16B-aligned chunks
  __shared__ bf16 sB[2][256][40];
  const int tid = threadIdx.x;
  const int wave = tid >> 5, lane = tid & 31;
  const int lm = lane & 15, lh = lane >> 4;
  const int bm = blockIdx.y * 128, bn = blockIdx.x * 256;
  const int wm = (wave & 1) * 64, wn = (wave >> 1) * 64;

  v8f zero8 = {0.f, 0.f, 0.f, 0.f, 0.f, 0.f, 0.f, 0.f};
  v8f acc[4][4];
#pragma unroll
  for (int i = 0; i < 4; ++i)
#pragma unroll
    for (int j = 0; j < 4; ++j) acc[i][j] = zero8;

  // staging assignments: A = 512 16B-chunks, B = 1024 16B-chunks
#if HAVE_ASYNC_LDS
  {
#pragma unroll
    for (int t = 0; t < 2; ++t) {
      int id = tid + t * 256, row = id >> 2, c = (id & 3) * 8;
      async_cp16(&sA[0][row][c], A + (size_t)(bm + row) * K + c);
    }
#pragma unroll
    for (int t = 0; t < 4; ++t) {
      int id = tid + t * 256, row = id >> 2, c = (id & 3) * 8;
      async_cp16(&sB[0][row][c], Bt + (size_t)(bn + row) * K + c);
    }
  }
#endif

  for (int k0 = 0, it = 0; k0 < K; k0 += 32, ++it) {
    const int buf = it & 1;
#if HAVE_ASYNC_LDS
    __builtin_amdgcn_s_wait_asynccnt(0);
    __syncthreads();
    if (k0 + 32 < K) {
#pragma unroll
      for (int t = 0; t < 2; ++t) {
        int id = tid + t * 256, row = id >> 2, c = (id & 3) * 8;
        async_cp16(&sA[buf ^ 1][row][c], A + (size_t)(bm + row) * K + k0 + 32 + c);
      }
#pragma unroll
      for (int t = 0; t < 4; ++t) {
        int id = tid + t * 256, row = id >> 2, c = (id & 3) * 8;
        async_cp16(&sB[buf ^ 1][row][c], Bt + (size_t)(bn + row) * K + k0 + 32 + c);
      }
    }
#else
    __syncthreads();
#pragma unroll
    for (int t = 0; t < 2; ++t) {
      int id = tid + t * 256, row = id >> 2, c = (id & 3) * 8;
      const bf16* ga = A + (size_t)(bm + row) * K + k0 + c;
      *(v8bf*)&sA[buf][row][c] = *(const v8bf*)ga;
      if (k0 + 32 < K) __builtin_prefetch(ga + 32, 0, 1);
    }
#pragma unroll
    for (int t = 0; t < 4; ++t) {
      int id = tid + t * 256, row = id >> 2, c = (id & 3) * 8;
      const bf16* gb = Bt + (size_t)(bn + row) * K + k0 + c;
      *(v8bf*)&sB[buf][row][c] = *(const v8bf*)gb;
      if (k0 + 32 < K) __builtin_prefetch(gb + 32, 0, 1);
    }
    __syncthreads();
#endif

    v16bf af[4], bfv[4];
#pragma unroll
    for (int i = 0; i < 4; ++i) {
      const bf16* p = &sA[buf][wm + i * 16 + lm][0];
      af[i] = loadAB(p + 8 * lh, p + 16 + 8 * lh);       // A 16x32 layout
    }
#pragma unroll
    for (int j = 0; j < 4; ++j) {
      const bf16* p = &sB[buf][wn + j * 16 + lm][0];
      bfv[j] = loadAB(p + 8 * lh, p + 16 + 8 * lh);      // B 32x16 layout
    }
#pragma unroll
    for (int i = 0; i < 4; ++i)
#pragma unroll
      for (int j = 0; j < 4; ++j) acc[i][j] = wmma_bf16(af[i], bfv[j], acc[i][j]);

#if HAVE_ASYNC_LDS
    __syncthreads();   // all waves done reading buf before it is refilled
#endif
  }

#pragma unroll
  for (int i = 0; i < 4; ++i)
#pragma unroll
    for (int j = 0; j < 4; ++j)
#pragma unroll
      for (int r = 0; r < 8; ++r) {
        int gr = bm + wm + i * 16 + lh * 8 + r;
        int gc = bn + wn + j * 16 + lm;
        size_t idx = (size_t)gr * N + gc;
        float v = acc[i][j][r];
        if (MODE == 2 || MODE == 3) v += bias[gc];
        if (MODE == 2) v = fmaxf(v, 0.f);
        if (MODE == 1 || MODE == 3) v += resid[idx];
        if (MODE == 0 || MODE == 2) Cb[idx] = (bf16)v;
        else                         Cf[idx] = v;
      }
}

// ---------------------------------------------------------------- fused flash attention
__global__ __launch_bounds__(128) void attn_kernel(
    const bf16* __restrict__ Qb, const bf16* __restrict__ Kb,
    const bf16* __restrict__ Vb, const unsigned long long* __restrict__ mb,
    bf16* __restrict__ Ob) {
  __shared__ bf16 sK[64][72];        // [key][dk]
  __shared__ bf16 sVt[64][72];       // [dv][key]
  __shared__ bf16 sP[4][16][72];     // per-wave P staging
  const int tid = threadIdx.x;
  const int wave = tid >> 5, lane = tid & 31;
  const int lm = lane & 15, lh = lane >> 4;
  const int bh = blockIdx.y;
  const int b = bh >> 3, hh = bh & 7;
  const int q0 = blockIdx.x * 64 + wave * 16;
  const float scale = 0.125f;

  v16bf qf[2];
  {
    const bf16* qrow = Qb + ((size_t)(b * NN + q0 + lm) * HH + hh) * DK;
#pragma unroll
    for (int t = 0; t < 2; ++t)
      qf[t] = loadAB(qrow + t * 32 + 8 * lh, qrow + t * 32 + 16 + 8 * lh);
  }

  v8f zero8 = {0.f, 0.f, 0.f, 0.f, 0.f, 0.f, 0.f, 0.f};
  v8f acc[4];
#pragma unroll
  for (int jf = 0; jf < 4; ++jf) acc[jf] = zero8;
  float mrun[8], lrun[8];
#pragma unroll
  for (int r = 0; r < 8; ++r) { mrun[r] = -__builtin_inff(); lrun[r] = 0.f; }

  for (int kb = 0; kb < NN; kb += 64) {
    __syncthreads();
#pragma unroll
    for (int t = 0; t < 4; ++t) {                 // stage K block
      int id = tid + t * 128;
      int key = id >> 3, c = (id & 7) * 8;
      *(v8bf*)&sK[key][c] =
          *(const v8bf*)(Kb + ((size_t)(b * NN + kb + key) * HH + hh) * DK + c);
    }
#pragma unroll
    for (int t = 0; t < 4; ++t) {                 // stage V transposed
      int id = tid + t * 128;
      int key = id >> 3, c = (id & 7) * 8;
      v8bf v = *(const v8bf*)(Vb + ((size_t)(b * NN + kb + key) * HH + hh) * DK + c);
#pragma unroll
      for (int j = 0; j < 8; ++j) sVt[c + j][key] = v[j];
    }
    __syncthreads();

    // S = Q K^T
    v8f sf[4];
#pragma unroll
    for (int kg = 0; kg < 4; ++kg) {
      const bf16* kr = &sK[kg * 16 + lm][0];
      v16bf k0 = loadAB(kr + 8 * lh, kr + 16 + 8 * lh);
      v16bf k1 = loadAB(kr + 32 + 8 * lh, kr + 32 + 16 + 8 * lh);
      v8f z = zero8;
      z = wmma_bf16(qf[0], k0, z);
      z = wmma_bf16(qf[1], k1, z);
      sf[kg] = z;
    }
    // scale + mask via packed bits (one u64 covers keys kb..kb+63 of a row)
#pragma unroll
    for (int r = 0; r < 8; ++r) {
      int mq = q0 + r + 8 * lh;
      unsigned long long w = mb[((size_t)(b * NN + mq)) * 8 + (kb >> 6)];
#pragma unroll
      for (int kg = 0; kg < 4; ++kg) {
        float x = sf[kg][r] * scale;
        if ((w >> (kg * 16 + lm)) & 1ull) x = -__builtin_inff();
        sf[kg][r] = x;
      }
    }
    // online softmax
#pragma unroll
    for (int r = 0; r < 8; ++r) {
      float mx = fmaxf(fmaxf(sf[0][r], sf[1][r]), fmaxf(sf[2][r], sf[3][r]));
#pragma unroll
      for (int s = 1; s < 16; s <<= 1) mx = fmaxf(mx, __shfl_xor(mx, s, 16));
      float mn = fmaxf(mrun[r], mx);
      float alpha = __expf(mrun[r] - mn);
      mrun[r] = mn;
      float rs = 0.f;
#pragma unroll
      for (int kg = 0; kg < 4; ++kg) {
        float p = __expf(sf[kg][r] - mn);
        sf[kg][r] = p;
        rs += p;
      }
#pragma unroll
      for (int s = 1; s < 16; s <<= 1) rs += __shfl_xor(rs, s, 16);
      lrun[r] = lrun[r] * alpha + rs;
#pragma unroll
      for (int jf = 0; jf < 4; ++jf) acc[jf][r] = acc[jf][r] * alpha;
    }
    // P (C layout) -> LDS -> A-fragment layout (same-wave DS, in order)
#pragma unroll
    for (int kg = 0; kg < 4; ++kg)
#pragma unroll
      for (int r = 0; r < 8; ++r)
        sP[wave][r + 8 * lh][kg * 16 + lm] = (bf16)sf[kg][r];
    v16bf pf[2];
    {
      const bf16* pr = &sP[wave][lm][0];
#pragma unroll
      for (int t = 0; t < 2; ++t)
        pf[t] = loadAB(pr + t * 32 + 8 * lh, pr + t * 32 + 16 + 8 * lh);
    }
    // O += P * V
#pragma unroll
    for (int jf = 0; jf < 4; ++jf) {
      const bf16* vr = &sVt[jf * 16 + lm][0];
      v16bf v0 = loadAB(vr + 8 * lh, vr + 16 + 8 * lh);
      v16bf v1 = loadAB(vr + 32 + 8 * lh, vr + 32 + 16 + 8 * lh);
      acc[jf] = wmma_bf16(pf[0], v0, acc[jf]);
      acc[jf] = wmma_bf16(pf[1], v1, acc[jf]);
    }
  }
#pragma unroll
  for (int jf = 0; jf < 4; ++jf)
#pragma unroll
    for (int r = 0; r < 8; ++r) {
      int mq = q0 + r + 8 * lh;
      float o = acc[jf][r] / lrun[r];
      Ob[((size_t)(b * NN + mq) * HH + hh) * DK + jf * 16 + lm] = (bf16)o;
    }
}

// ---------------------------------------------------------------- BatchNorm (training mode, biased var), 4 cols/block
__global__ __launch_bounds__(256) void bn_stats(const float* __restrict__ X,
                                                float* __restrict__ mu,
                                                float* __restrict__ rinv) {
  __shared__ float ss[4][256], sq[4][256];
  int d0 = blockIdx.x * 4;
  float s[4] = {0.f, 0.f, 0.f, 0.f}, q[4] = {0.f, 0.f, 0.f, 0.f};
  for (int r = threadIdx.x; r < (int)HB; r += 256) {
    const float4 v = *(const float4*)&X[(size_t)r * DD + d0];
    s[0] += v.x; q[0] += v.x * v.x;
    s[1] += v.y; q[1] += v.y * v.y;
    s[2] += v.z; q[2] += v.z * v.z;
    s[3] += v.w; q[3] += v.w * v.w;
  }
#pragma unroll
  for (int c = 0; c < 4; ++c) { ss[c][threadIdx.x] = s[c]; sq[c][threadIdx.x] = q[c]; }
  __syncthreads();
  for (int st = 128; st > 0; st >>= 1) {
    if (threadIdx.x < st) {
#pragma unroll
      for (int c = 0; c < 4; ++c) {
        ss[c][threadIdx.x] += ss[c][threadIdx.x + st];
        sq[c][threadIdx.x] += sq[c][threadIdx.x + st];
      }
    }
    __syncthreads();
  }
  if (threadIdx.x < 4) {
    float m = ss[threadIdx.x][0] * (1.f / (float)HB);
    float var = sq[threadIdx.x][0] * (1.f / (float)HB) - m * m;
    mu[d0 + threadIdx.x] = m;
    rinv[d0 + threadIdx.x] = rsqrtf(var + 1e-5f);
  }
}

__global__ void bn_apply(const float* __restrict__ X, const float* __restrict__ mu,
                         const float* __restrict__ rinv, const float* __restrict__ g,
                         const float* __restrict__ bt, float* __restrict__ H,
                         bf16* __restrict__ Hb) {
  size_t i4 = ((size_t)blockIdx.x * 256 + threadIdx.x) * 4;
  int d = (int)(i4 & 511);
  const float4 xv = *(const float4*)&X[i4];
  const float4 m = *(const float4*)&mu[d];
  const float4 ri = *(const float4*)&rinv[d];
  const float4 gg = *(const float4*)&g[d];
  const float4 bb = *(const float4*)&bt[d];
  float4 y;
  y.x = (xv.x - m.x) * ri.x * gg.x + bb.x;
  y.y = (xv.y - m.y) * ri.y * gg.y + bb.y;
  y.z = (xv.z - m.z) * ri.z * gg.z + bb.z;
  y.w = (xv.w - m.w) * ri.w * gg.w + bb.w;
  *(float4*)&H[i4] = y;
  v4bf o = {(bf16)y.x, (bf16)y.y, (bf16)y.z, (bf16)y.w};
  *(v4bf*)&Hb[i4] = o;
}

// ---------------------------------------------------------------- output: (h, h.mean(axis=1))
__global__ void copy_out(const float* __restrict__ H, float* __restrict__ out) {
  size_t i4 = ((size_t)blockIdx.x * 256 + threadIdx.x) * 4;
  *(float4*)&out[i4] = *(const float4*)&H[i4];
}
__global__ void mean_out(const float* __restrict__ H, float* __restrict__ out) {
  int i = blockIdx.x * 256 + threadIdx.x;   // B*D/4 = 4096
  int b = i >> 7, d4 = (i & 127) * 4;
  float4 s = {0.f, 0.f, 0.f, 0.f};
  for (int n = 0; n < NN; ++n) {
    const float4 v = *(const float4*)&H[((size_t)(b * NN + n)) * DD + d4];
    s.x += v.x; s.y += v.y; s.z += v.z; s.w += v.w;
  }
  s.x *= (1.f / (float)NN); s.y *= (1.f / (float)NN);
  s.z *= (1.f / (float)NN); s.w *= (1.f / (float)NN);
  *(float4*)&out[b * DD + d4] = s;
}

// ---------------------------------------------------------------- launcher
extern "C" void kernel_launch(void* const* d_in, const int* in_sizes, int n_in,
                              void* d_out, int out_size, void* d_ws, size_t ws_size,
                              hipStream_t stream) {
  (void)in_sizes; (void)n_in; (void)out_size; (void)ws_size;
  const float* x    = (const float*)d_in[0];
  const int*   mask = (const int*)d_in[1];
  const float* We   = (const float*)d_in[2];
  const float* be   = (const float*)d_in[3];
  const float* Wq   = (const float*)d_in[4];
  const float* Wk   = (const float*)d_in[5];
  const float* Wv   = (const float*)d_in[6];
  const float* Wo   = (const float*)d_in[7];
  const float* g1   = (const float*)d_in[8];
  const float* bt1  = (const float*)d_in[9];
  const float* g2   = (const float*)d_in[10];
  const float* bt2  = (const float*)d_in[11];
  const float* W1   = (const float*)d_in[12];
  const float* bf1  = (const float*)d_in[13];
  const float* W2   = (const float*)d_in[14];
  const float* bf2  = (const float*)d_in[15];

  char* ws = (char*)d_ws;
  float* h     = (float*)(ws + OFF_H);
  float* hres  = (float*)(ws + OFF_HRES);
  bf16*  hbf   = (bf16*)(ws + OFF_HBF);
  bf16*  qbf   = (bf16*)(ws + OFF_QBF);
  bf16*  kbf   = (bf16*)(ws + OFF_KBF);
  bf16*  vbf   = (bf16*)(ws + OFF_VBF);
  bf16*  heads = (bf16*)(ws + OFF_HEADS);
  bf16*  ffbf  = (bf16*)(ws + OFF_FFBF);
  bf16*  wqT   = (bf16*)(ws + OFF_WQT);
  bf16*  wkT   = (bf16*)(ws + OFF_WKT);
  bf16*  wvT   = (bf16*)(ws + OFF_WVT);
  bf16*  woT   = (bf16*)(ws + OFF_WOT);
  bf16*  w1T   = (bf16*)(ws + OFF_W1T);
  bf16*  w2T   = (bf16*)(ws + OFF_W2T);
  float* mu    = (float*)(ws + OFF_MU);
  float* rinv  = (float*)(ws + OFF_RINV);
  unsigned long long* mbits = (unsigned long long*)(ws + OFF_MB);

  const int elemHD = (int)(HB * DD);          // 8388608
  init_embed<<<elemHD / 1024, 256, 0, stream>>>(x, We, be, h, hbf);
  pack_mask<<<(BB * NN * 8) / 256, 256, 0, stream>>>(mask, mbits);

  const dim3 gemmGrid512(2, 128), gemmGrid2048(8, 128);
  const size_t wLayerQKV = (size_t)HH * DD * DK;    // 262144
  const size_t wLayerFF  = (size_t)DD * FF;         // 1048576

  for (int l = 0; l < LL; ++l) {
    pack_qkvT<<<262144 / 256, 256, 0, stream>>>(Wq + l * wLayerQKV, wqT);
    pack_qkvT<<<262144 / 256, 256, 0, stream>>>(Wk + l * wLayerQKV, wkT);
    pack_qkvT<<<262144 / 256, 256, 0, stream>>>(Wv + l * wLayerQKV, wvT);
    pack_woT <<<262144 / 256, 256, 0, stream>>>(Wo + l * wLayerQKV, woT);
    pack_w1T <<<1048576 / 256, 256, 0, stream>>>(W1 + l * wLayerFF, w1T);
    pack_w2T <<<1048576 / 256, 256, 0, stream>>>(W2 + l * wLayerFF, w2T);

    gemm_bf16<0><<<gemmGrid512, 256, 0, stream>>>(hbf, wqT, nullptr, nullptr, nullptr, qbf, (int)HB, DD, DD);
    gemm_bf16<0><<<gemmGrid512, 256, 0, stream>>>(hbf, wkT, nullptr, nullptr, nullptr, kbf, (int)HB, DD, DD);
    gemm_bf16<0><<<gemmGrid512, 256, 0, stream>>>(hbf, wvT, nullptr, nullptr, nullptr, vbf, (int)HB, DD, DD);

    attn_kernel<<<dim3(NN / 64, BB * HH), 128, 0, stream>>>(qbf, kbf, vbf, mbits, heads);

    gemm_bf16<1><<<gemmGrid512, 256, 0, stream>>>(heads, woT, nullptr, h, hres, nullptr, (int)HB, DD, DD);
    bn_stats<<<DD / 4, 256, 0, stream>>>(hres, mu, rinv);
    bn_apply<<<elemHD / 1024, 256, 0, stream>>>(hres, mu, rinv, g1 + l * DD, bt1 + l * DD, h, hbf);

    gemm_bf16<2><<<gemmGrid2048, 256, 0, stream>>>(hbf, w1T, bf1 + l * FF, nullptr, nullptr, ffbf, (int)HB, FF, DD);
    gemm_bf16<3><<<gemmGrid512, 256, 0, stream>>>(ffbf, w2T, bf2 + l * DD, h, hres, nullptr, (int)HB, DD, FF);
    bn_stats<<<DD / 4, 256, 0, stream>>>(hres, mu, rinv);
    bn_apply<<<elemHD / 1024, 256, 0, stream>>>(hres, mu, rinv, g2 + l * DD, bt2 + l * DD, h, hbf);
  }

  copy_out<<<elemHD / 1024, 256, 0, stream>>>(h, (float*)d_out);
  mean_out<<<(BB * DD / 4) / 256, 256, 0, stream>>>(h, (float*)d_out + HB * DD);
}
